// GraphAttentionNet_40278203302098
// MI455X (gfx1250) — compile-verified, matrix operand order checked by reading, not run on hardware
//
#include <hip/hip_runtime.h>
#include <hip/hip_bf16.h>
#include <math.h>

// ---------------------------------------------------------------------------
// GAT (2-layer) for MI455X / gfx1250.  Dense feature transforms run on the
// bf16 WMMA pipe (v_wmma_f32_16x16x32_bf16, f32 accum); sparse softmax /
// scatter phases use L2-resident gathers + atomics.
// ---------------------------------------------------------------------------

typedef __attribute__((ext_vector_type(16))) __bf16     v16bf;
typedef __attribute__((ext_vector_type(8)))  float      v8f;
typedef __attribute__((ext_vector_type(4)))  unsigned   u32x4;

#define LRELU_SLOPE 0.2f
#define LN_EPS      1e-5f

static __device__ __forceinline__ unsigned short f2bf(float f) {
  unsigned u = __float_as_uint(f);
  u += 0x7FFFu + ((u >> 16) & 1u);        // round-to-nearest-even
  return (unsigned short)(u >> 16);
}
// order-preserving float<->int mapping for atomicMax on floats
static __device__ __forceinline__ int f2ord(float f) {
  int i = __float_as_int(f);
  return i >= 0 ? i : (i ^ 0x7FFFFFFF);
}
static __device__ __forceinline__ float ord2f(int i) {
  return __int_as_float(i >= 0 ? i : (i ^ 0x7FFFFFFF));
}

// ---------------------------------------------------------------------------
// utility fills / conversions
// ---------------------------------------------------------------------------
__global__ void fill_f32_k(float* p, float v, long long n) {
  for (long long t = blockIdx.x * (long long)blockDim.x + threadIdx.x; t < n;
       t += (long long)gridDim.x * blockDim.x)
    p[t] = v;
}
__global__ void fill_i32_k(int* p, int v, long long n) {
  for (long long t = blockIdx.x * (long long)blockDim.x + threadIdx.x; t < n;
       t += (long long)gridDim.x * blockDim.x)
    p[t] = v;
}
__global__ void f32_to_bf16_k(const float* __restrict__ s,
                              unsigned short* __restrict__ d, long long n) {
  for (long long t = blockIdx.x * (long long)blockDim.x + threadIdx.x; t < n;
       t += (long long)gridDim.x * blockDim.x)
    d[t] = f2bf(s[t]);
}
// W [K,Nc] row-major f32  ->  Wt [Nc,K] row-major bf16 (i.e. W^T)
__global__ void transpose_w_bf16_k(const float* __restrict__ W,
                                   unsigned short* __restrict__ Wt,
                                   int K, int Nc) {
  int tot = K * Nc;
  for (int t = blockIdx.x * blockDim.x + threadIdx.x; t < tot;
       t += gridDim.x * blockDim.x) {
    int k = t / Nc, n = t - k * Nc;
    Wt[(size_t)n * K + k] = f2bf(W[t]);
  }
}

// ---------------------------------------------------------------------------
// bf16 WMMA GEMM:  C[M,Nc] = A[M,K] * Bt[Nc,K]^T   (f32 accumulate)
// one wave -> 16 rows x 64 cols (4 accumulators share one A fragment).
// K / NC are compile-time so the k-loop fully unrolls and loads pipeline
// across the WMMAs.  A (activations) is streamed once -> non-temporal, so
// L2 stays reserved for the feature tables the edge phases gather from.
// launch: grid = M/16 blocks, blockDim = 32 * (NC/64)
// ---------------------------------------------------------------------------
template <int K, int NC>
__global__ void gemm_bf16_wmma_k(const unsigned short* __restrict__ A,
                                 const unsigned short* __restrict__ Bt,
                                 float* __restrict__ C) {
  const int wave  = threadIdx.x >> 5;
  const int lane  = threadIdx.x & 31;
  const int l16   = lane & 15;
  const int hi    = lane >> 4;          // 0: K 0-7/16-23  1: K 8-15/24-31
  const int mBase = blockIdx.x << 4;
  const int nBase = wave << 6;

  v8f acc[4] = {};

  const unsigned short* arow = A + (size_t)(mBase + l16) * K;
#pragma unroll
  for (int k0 = 0; k0 < K; k0 += 32) {
    union { v16bf v; u32x4 q[2]; } a;
    a.q[0] = __builtin_nontemporal_load((const u32x4*)(arow + k0 + hi * 8));
    a.q[1] = __builtin_nontemporal_load((const u32x4*)(arow + k0 + 16 + hi * 8));
#pragma unroll
    for (int j = 0; j < 4; ++j) {
      const unsigned short* brow = Bt + (size_t)(nBase + j * 16 + l16) * K;
      union { v16bf v; u32x4 q[2]; } b;
      b.q[0] = *(const u32x4*)(brow + k0 + hi * 8);
      b.q[1] = *(const u32x4*)(brow + k0 + 16 + hi * 8);
      acc[j] = __builtin_amdgcn_wmma_f32_16x16x32_bf16(
          false, a.v, false, b.v, (short)0, acc[j], false, false);
    }
  }
#pragma unroll
  for (int j = 0; j < 4; ++j) {
    const int col = nBase + j * 16 + l16;
#pragma unroll
    for (int i = 0; i < 8; ++i)
      C[(size_t)(mBase + hi * 8 + i) * NC + col] = acc[j][i];
  }
}

// ---------------------------------------------------------------------------
// attention logits: al_s[n,h] = <H[n,h,:], a_src[h,:]> ; same for dst
// ---------------------------------------------------------------------------
__global__ void att_logits_k(const float* __restrict__ H,
                             const float* __restrict__ a_src,
                             const float* __restrict__ a_dst,
                             float* __restrict__ al_s,
                             float* __restrict__ al_d,
                             int Nn, int heads) {
  int tot = Nn * heads;
  for (int t = blockIdx.x * blockDim.x + threadIdx.x; t < tot;
       t += gridDim.x * blockDim.x) {
    int n = t / heads, h = t - n * heads;
    const float* row = H + (size_t)n * heads * 64 + h * 64;
    const float* as  = a_src + h * 64;
    const float* ad  = a_dst + h * 64;
    float s = 0.f, d = 0.f;
#pragma unroll 8
    for (int c = 0; c < 64; ++c) { s += row[c] * as[c]; d += row[c] * ad[c]; }
    al_s[t] = s;
    al_d[t] = d;
  }
}

// ---------------------------------------------------------------------------
// edge phase A: segment max of leaky_relu(al_s[src]+al_d[dst]) over dst
// ---------------------------------------------------------------------------
__global__ void edge_max_k(const long long* __restrict__ ei,
                           const float* __restrict__ al_s,
                           const float* __restrict__ al_d,
                           int* __restrict__ m,
                           int E, int Nn, int heads) {
  long long tot = (long long)(E + Nn) * heads;
  for (long long t = blockIdx.x * (long long)blockDim.x + threadIdx.x; t < tot;
       t += (long long)gridDim.x * blockDim.x) {
    int e = (int)(t / heads), h = (int)(t - (long long)e * heads);
    long long s, d;
    if (e < E) { s = ei[e]; d = ei[E + e]; } else { s = d = e - E; }
    float v = al_s[s * heads + h] + al_d[d * heads + h];
    v = v > 0.f ? v : LRELU_SLOPE * v;
    atomicMax(&m[d * heads + h], f2ord(v));
  }
}

// ---------------------------------------------------------------------------
// edge phase B: eexp = exp(e - max[dst]); denom[dst] += eexp
// ---------------------------------------------------------------------------
__global__ void edge_expsum_k(const long long* __restrict__ ei,
                              const float* __restrict__ al_s,
                              const float* __restrict__ al_d,
                              const int* __restrict__ m,
                              float* __restrict__ eexp,
                              float* __restrict__ den,
                              int E, int Nn, int heads) {
  long long tot = (long long)(E + Nn) * heads;
  for (long long t = blockIdx.x * (long long)blockDim.x + threadIdx.x; t < tot;
       t += (long long)gridDim.x * blockDim.x) {
    int e = (int)(t / heads), h = (int)(t - (long long)e * heads);
    long long s, d;
    if (e < E) { s = ei[e]; d = ei[E + e]; } else { s = d = e - E; }
    float v = al_s[s * heads + h] + al_d[d * heads + h];
    v = v > 0.f ? v : LRELU_SLOPE * v;
    float ee = __expf(v - ord2f(m[d * heads + h]));
    eexp[t] = ee;
    atomicAdd(&den[d * heads + h], ee);
  }
}

// ---------------------------------------------------------------------------
// edge phase C: ACC[dst,h,:] += H[src,h,:] * (eexp/den[dst,h])
// one wave per (edge, head); lanes cover the 64 channels (2 each)
// ---------------------------------------------------------------------------
__global__ void edge_msg_k(const long long* __restrict__ ei,
                           const float* __restrict__ H,
                           const float* __restrict__ eexp,
                           const float* __restrict__ den,
                           float* __restrict__ ACC,
                           int E, int Nn, int heads) {
  int lane = threadIdx.x & 31;
  long long w0   = (blockIdx.x * (long long)blockDim.x + threadIdx.x) >> 5;
  long long step = ((long long)gridDim.x * blockDim.x) >> 5;
  long long tot  = (long long)(E + Nn) * heads;
  int ldH = heads * 64;
  for (long long w = w0; w < tot; w += step) {
    int e = (int)(w / heads), h = (int)(w - (long long)e * heads);
    long long s, d;
    if (e < E) { s = ei[e]; d = ei[E + e]; } else { s = d = e - E; }
    float alpha = eexp[w] / den[d * heads + h];
    const float* hs = H + (size_t)s * ldH + h * 64;
    float*       od = ACC + (size_t)d * ldH + h * 64;
    atomicAdd(&od[lane],      hs[lane]      * alpha);
    atomicAdd(&od[lane + 32], hs[lane + 32] * alpha);
  }
}

// ---------------------------------------------------------------------------
// z = elu(layernorm(ACC + b1)); Z (f32, in place) and Zb (bf16 for GEMM2)
// one wave per node, 256 channels
// ---------------------------------------------------------------------------
__global__ void ln_elu_k(float* __restrict__ Z,
                         unsigned short* __restrict__ Zb,
                         const float* __restrict__ b1,
                         const float* __restrict__ lw,
                         const float* __restrict__ lb,
                         int Nn) {
  int lane = threadIdx.x & 31;
  int w0   = (blockIdx.x * blockDim.x + threadIdx.x) >> 5;
  int step = (gridDim.x * blockDim.x) >> 5;
  for (int n = w0; n < Nn; n += step) {
    float* row = Z + (size_t)n * 256;
    float v[8];
    float s = 0.f, ss = 0.f;
#pragma unroll
    for (int t = 0; t < 8; ++t) {
      int c = lane + 32 * t;
      float x = row[c] + b1[c];
      v[t] = x; s += x; ss += x * x;
    }
#pragma unroll
    for (int off = 16; off > 0; off >>= 1) {
      s  += __shfl_xor(s, off, 32);
      ss += __shfl_xor(ss, off, 32);
    }
    float mu  = s * (1.0f / 256.0f);
    float var = ss * (1.0f / 256.0f) - mu * mu;
    float rs  = rsqrtf(var + LN_EPS);
#pragma unroll
    for (int t = 0; t < 8; ++t) {
      int c = lane + 32 * t;
      float y = (v[t] - mu) * rs * lw[c] + lb[c];
      y = y > 0.f ? y : (__expf(y) - 1.f);   // ELU
      row[c] = y;
      Zb[(size_t)n * 256 + c] = f2bf(y);
    }
  }
}

// ---------------------------------------------------------------------------
// out[n,c] = mean over 2 heads of ACC2[n,h,c] + b2[c]
// ---------------------------------------------------------------------------
__global__ void final_k(const float* __restrict__ ACC2,
                        const float* __restrict__ b2,
                        float* __restrict__ out, int Nn) {
  int tot = Nn * 64;
  for (int t = blockIdx.x * blockDim.x + threadIdx.x; t < tot;
       t += gridDim.x * blockDim.x) {
    int n = t >> 6, c = t & 63;
    out[t] = 0.5f * (ACC2[(size_t)n * 128 + c] + ACC2[(size_t)n * 128 + 64 + c])
             + b2[c];
  }
}

// ---------------------------------------------------------------------------
// launch
// ---------------------------------------------------------------------------
extern "C" void kernel_launch(void* const* d_in, const int* in_sizes, int n_in,
                              void* d_out, int out_size, void* d_ws, size_t ws_size,
                              hipStream_t stream) {
  (void)in_sizes; (void)n_in; (void)out_size; (void)ws_size;
  constexpr int Nn = 50000, Ee = 800000;
  constexpr int H1 = 4, H2 = 2;

  const float*     x    = (const float*)d_in[0];
  const long long* ei   = (const long long*)d_in[1];  // int64 [2,E]
  const float*     W1   = (const float*)d_in[2];      // [128,256]
  const float*     as1  = (const float*)d_in[3];
  const float*     ad1  = (const float*)d_in[4];
  const float*     b1   = (const float*)d_in[5];
  const float*     lnw  = (const float*)d_in[6];
  const float*     lnb  = (const float*)d_in[7];
  const float*     W2   = (const float*)d_in[8];      // [256,128]
  const float*     as2  = (const float*)d_in[9];
  const float*     ad2  = (const float*)d_in[10];
  const float*     b2   = (const float*)d_in[11];
  float*           out  = (float*)d_out;

  // ---- workspace layout (256-byte aligned slices) ----
  size_t off = 0;
  auto alloc = [&](size_t bytes) -> char* {
    char* p = (char*)d_ws + off;
    off += (bytes + 255) & ~(size_t)255;
    return p;
  };
  unsigned short* xb    = (unsigned short*)alloc((size_t)Nn * 128 * 2);
  unsigned short* w1t   = (unsigned short*)alloc((size_t)256 * 128 * 2);
  unsigned short* w2t   = (unsigned short*)alloc((size_t)128 * 256 * 2);
  float*          Hf    = (float*)alloc((size_t)Nn * 256 * 4);
  float*          ACC   = (float*)alloc((size_t)Nn * 256 * 4);  // then Z in place
  unsigned short* zb    = (unsigned short*)alloc((size_t)Nn * 256 * 2);
  float*          G     = (float*)alloc((size_t)Nn * 128 * 4);
  float*          ACC2  = (float*)alloc((size_t)Nn * 128 * 4);
  float*          alS1  = (float*)alloc((size_t)Nn * H1 * 4);
  float*          alD1  = (float*)alloc((size_t)Nn * H1 * 4);
  int*            m1    = (int*)alloc((size_t)Nn * H1 * 4);
  float*          den1  = (float*)alloc((size_t)Nn * H1 * 4);
  float*          eex1  = (float*)alloc((size_t)(Ee + Nn) * H1 * 4);
  float*          alS2  = (float*)alloc((size_t)Nn * H2 * 4);
  float*          alD2  = (float*)alloc((size_t)Nn * H2 * 4);
  int*            m2    = (int*)alloc((size_t)Nn * H2 * 4);
  float*          den2  = (float*)alloc((size_t)Nn * H2 * 4);
  float*          eex2  = (float*)alloc((size_t)(Ee + Nn) * H2 * 4);

  const int GS_BLK = 2048, GS_THR = 256;
  const int NEG_ORD = (int)0x80800000u;   // f2ord(-FLT_MAX)

  // ---- init / conversion ----
  f32_to_bf16_k<<<GS_BLK, GS_THR, 0, stream>>>(x, xb, (long long)Nn * 128);
  transpose_w_bf16_k<<<256, 256, 0, stream>>>(W1, w1t, 128, 256);
  transpose_w_bf16_k<<<256, 256, 0, stream>>>(W2, w2t, 256, 128);
  fill_f32_k<<<GS_BLK, GS_THR, 0, stream>>>(ACC,  0.f, (long long)Nn * 256);
  fill_f32_k<<<GS_BLK, GS_THR, 0, stream>>>(ACC2, 0.f, (long long)Nn * 128);
  fill_f32_k<<<64, 256, 0, stream>>>(den1, 0.f, (long long)Nn * H1);
  fill_f32_k<<<64, 256, 0, stream>>>(den2, 0.f, (long long)Nn * H2);
  fill_i32_k<<<64, 256, 0, stream>>>(m1, NEG_ORD, (long long)Nn * H1);
  fill_i32_k<<<64, 256, 0, stream>>>(m2, NEG_ORD, (long long)Nn * H2);

  // ---- layer 1 ----
  gemm_bf16_wmma_k<128, 256><<<Nn / 16, 128, 0, stream>>>(xb, w1t, Hf);
  att_logits_k<<<1024, 256, 0, stream>>>(Hf, as1, ad1, alS1, alD1, Nn, H1);
  edge_max_k<<<GS_BLK, GS_THR, 0, stream>>>(ei, alS1, alD1, m1, Ee, Nn, H1);
  edge_expsum_k<<<GS_BLK, GS_THR, 0, stream>>>(ei, alS1, alD1, m1, eex1, den1,
                                               Ee, Nn, H1);
  edge_msg_k<<<GS_BLK, GS_THR, 0, stream>>>(ei, Hf, eex1, den1, ACC, Ee, Nn, H1);
  ln_elu_k<<<GS_BLK, GS_THR, 0, stream>>>(ACC, zb, b1, lnw, lnb, Nn);

  // ---- layer 2 ----
  gemm_bf16_wmma_k<256, 128><<<Nn / 16, 64, 0, stream>>>(zb, w2t, G);
  att_logits_k<<<1024, 256, 0, stream>>>(G, as2, ad2, alS2, alD2, Nn, H2);
  edge_max_k<<<GS_BLK, GS_THR, 0, stream>>>(ei, alS2, alD2, m2, Ee, Nn, H2);
  edge_expsum_k<<<GS_BLK, GS_THR, 0, stream>>>(ei, alS2, alD2, m2, eex2, den2,
                                               Ee, Nn, H2);
  edge_msg_k<<<GS_BLK, GS_THR, 0, stream>>>(ei, G, eex2, den2, ACC2, Ee, Nn, H2);
  final_k<<<GS_BLK, GS_THR, 0, stream>>>(ACC2, b2, out, Nn);
}
